// DynLLM_66185446031742
// MI455X (gfx1250) — matrix operand order repaired and checked by reading, not compiled, not run on hardware
//
// CDNA5 / gfx1250 implementation of the recurrent dynamical LM.
//
// Roofline: phase-3 output GEMM dominates (67 GFLOP, 262MB fp32 out -> ~11us
// HBM floor @23.3TB/s). Recurrence is latency-bound: one 1024-thread WG
// (32 waves) carries all B=16 batch lanes so every matmul is [16,512]x[512,N]
// -> one v_wmma_f32_16x16x32_bf16 16x16 tile per wave per 32-K chunk.
// Weights are pre-packed into per-lane WMMA B-fragment order (contiguous 32B
// loads per lane). Hebbian state is exploited analytically: only 32 rows of
// Hm are ever non-zero, so a 512KB bf16 buffer + closed-form softmax
// denominator replaces the 4MB/step scan.
//
// gfx1250-specific paths used:
//  - v_wmma_f32_16x16x32_bf16 for every matmul
//  - global_load_async_to_lds_b128 + s_wait_asynccnt (inline asm) for A-tile
//    staging in the dominant GEMM and for the per-step drive_u preload
//  - global_prefetch_b8 for the streamed packed-B operand

#include <hip/hip_runtime.h>

typedef __bf16 bf16_t;
typedef bf16_t v16bf __attribute__((ext_vector_type(16)));
typedef float v8f __attribute__((ext_vector_type(8)));
typedef unsigned short us8 __attribute__((ext_vector_type(8)));
typedef unsigned short us16 __attribute__((ext_vector_type(16)));

#define B_ 16
#define S_ 128
#define D_ 512
#define R_ 128
#define V_ 32000
#define NSLOT_ 8
#define HBROWS_ 32   /* only 32 Hebbian rows are ever written (S/4) */
#define CW_ 16
#define NH_ 4
#define DH_ 128

__device__ __forceinline__ unsigned short f2bf(float f) {
    union { float f; unsigned int u; } c; c.f = f;
    unsigned int u = c.u;
    u += 0x7fffu + ((u >> 16) & 1u);   // round-to-nearest-even
    return (unsigned short)(u >> 16);
}
__device__ __forceinline__ float bf2f(unsigned short h) {
    union { unsigned int u; float f; } c; c.u = ((unsigned int)h) << 16;
    return c.f;
}

// ---- async global->LDS copy (CDNA5 TDM-class path, tracked by ASYNCcnt) ----
// VDST = LDS byte address (flat LDS pointer truncates to the LDS offset),
// VADDR = 64-bit global address, saddr = off.
__device__ __forceinline__ void async_g2l_b128(void* lds, const void* gsrc) {
    unsigned ldsa = (unsigned)(unsigned long long)lds;
    asm volatile("global_load_async_to_lds_b128 %0, %1, off"
                 :: "v"(ldsa), "v"(gsrc) : "memory");
}
__device__ __forceinline__ void wait_async0() {
    asm volatile("s_wait_asynccnt 0x0" ::: "memory");
}

// ---------------------------------------------------------------------------
// Pack fp32 weight (logical K x N, optionally stored transposed as [N][K])
// into WMMA bf16 B-fragment order: tiles [ntile][ktile], within a tile
// lane-major [32 lanes][16 bf16].  B layout (16-bit, 32x16):
//   lanes 0-15  = col n, K {0..7,16..23} ; lanes 16-31 = col n, K {8..15,24..31}.
// ---------------------------------------------------------------------------
__global__ void pack_b_kernel(const float* __restrict__ src,
                              unsigned short* __restrict__ dst,
                              int K, int N, int trans) {
    const size_t total = (size_t)K * N;
    const int nkt = K >> 5;
    for (size_t i = (size_t)blockIdx.x * blockDim.x + threadIdx.x; i < total;
         i += (size_t)gridDim.x * blockDim.x) {
        int k = (int)(i / N);
        int n = (int)(i - (size_t)k * N);
        float v = trans ? src[(size_t)n * K + k] : src[i];
        int kt = k >> 5, kin = k & 31, nt = n >> 4, nin = n & 15;
        int lane = nin + (((kin >> 3) & 1) << 4);
        int e = (kin & 7) + ((kin >> 4) << 3);
        dst[((size_t)nt * nkt + kt) * 512 + lane * 16 + e] = f2bf(v);
    }
}

// ---------------------------------------------------------------------------
// One wave computes one 16x16 f32 tile of A[16xK] @ Bpacked[KxN] (ntile given).
// A: bf16 row-major in LDS (pitch elements). Per ISA A layout (16-bit 16x32):
//   lanes 0-15: M=lane,  K {0..7,16..23}; lanes 16-31: M=lane-16, K {8..15,24..31}
// Software-pipelined 1 deep so next fragments stay in flight under the WMMA.
// ---------------------------------------------------------------------------
__device__ __forceinline__ v8f wmma_acc_tile(const unsigned short* Alds,
                                             int apitch,
                                             const unsigned short* __restrict__ Bp,
                                             int K, int ntile, v8f acc) {
    const int lane = threadIdx.x & 31;
    const int row = lane & 15;
    const int hi = lane >> 4;
    const int nkt = K >> 5;
    const unsigned short* arow = Alds + row * apitch + hi * 8;
    const us16* bptr = (const us16*)(Bp + (size_t)ntile * nkt * 512 + lane * 16);
    us8 a0 = *(const us8*)(arow);
    us8 a1 = *(const us8*)(arow + 16);
    us16 bv = bptr[0];
    for (int kt = 0; kt < nkt; ++kt) {
        if (kt + 2 < nkt) __builtin_prefetch((const void*)(bptr + (kt + 2) * 32), 0, 1);
        const int nk = (kt + 1 < nkt) ? kt + 1 : kt;     // issue next frags first
        us8 na0 = *(const us8*)(arow + nk * 32);
        us8 na1 = *(const us8*)(arow + nk * 32 + 16);
        us16 nbv = bptr[nk * 32];                        // 512 ushorts per ktile
        us16 av = __builtin_shufflevector(a0, a1, 0, 1, 2, 3, 4, 5, 6, 7,
                                          8, 9, 10, 11, 12, 13, 14, 15);
        acc = __builtin_amdgcn_wmma_f32_16x16x32_bf16(
            false, __builtin_bit_cast(v16bf, av),
            false, __builtin_bit_cast(v16bf, bv),
            (short)0, acc, false, false);
        a0 = na0; a1 = na1; bv = nbv;
    }
    return acc;
}

// ---------------------------------------------------------------------------
// Shared GEMM kernel: C[m,n] = A[m,:] @ Bp + bias[n]   (fp32 out)
// Block: 256 threads = 8 waves as 4(m) x 2(n) -> block tile 64x32.
// A staged fully (64 rows x K bf16) in LDS: via ASYNC global->LDS copies when
// A is already bf16 (phase 3), via gather+convert when A = embed[ids] (phase 1).
// ---------------------------------------------------------------------------
__global__ __launch_bounds__(256) void gemm_bf16_kernel(
    const unsigned short* __restrict__ Abf, const int* __restrict__ ids,
    const float* __restrict__ embed, const unsigned short* __restrict__ Bp,
    const float* __restrict__ bias, float* __restrict__ C,
    int N, int K, int gather) {
    extern __shared__ char smem_raw[];
    unsigned short* Alds = (unsigned short*)smem_raw;   // 64 x K
    const int tid = threadIdx.x;
    const int m0 = blockIdx.x * 64;
    if (gather) {
        const int elems = 64 * K;
        for (int i = tid; i < elems; i += 256) {
            int r = i / K, k = i - r * K;
            int id = ids[m0 + r];
            Alds[i] = f2bf(embed[(size_t)id * K + k]);
        }
    } else {
        // 16B granules: 64 rows x (K/8) chunks, async straight into LDS.
        const int chunks = 64 * (K >> 3);
        for (int i = tid; i < chunks; i += 256) {
            int r = i >> 6, c = (i & 63) << 3;          // K=512 -> 64 chunks/row
            async_g2l_b128(Alds + r * K + c, Abf + (size_t)(m0 + r) * K + c);
        }
        wait_async0();
    }
    __syncthreads();
    const int wave = tid >> 5, lane = tid & 31;
    const int mi = wave >> 1, ni = wave & 1;
    const int ntg = blockIdx.y * 2 + ni;
    v8f acc = {0.f, 0.f, 0.f, 0.f, 0.f, 0.f, 0.f, 0.f};
    acc = wmma_acc_tile(Alds + mi * 16 * K, K, Bp, K, ntg, acc);
    // C/D layout: VGPR r: lanes0-15 -> M=r, N=lane; lanes16-31 -> M=r+8, N=lane-16
    const int cn = lane & 15, hi = lane >> 4;
    const int col = ntg * 16 + cn;
    const float bv = bias[col];
#pragma unroll
    for (int r = 0; r < 8; ++r) {
        int mrow = m0 + mi * 16 + r + hi * 8;
        C[(size_t)mrow * N + col] = acc[r] + bv;
    }
}

// ---------------------------------------------------------------------------
// Recurrence: single 1024-thread workgroup (32 wave32), all 16 batch lanes.
// ---------------------------------------------------------------------------
__global__ __launch_bounds__(1024) void recurrence_kernel(
    const unsigned short* __restrict__ WqP, const unsigned short* __restrict__ WkP,
    const unsigned short* __restrict__ WvP, const unsigned short* __restrict__ WoP,
    const unsigned short* __restrict__ WmemP, const unsigned short* __restrict__ WcP,
    const unsigned short* __restrict__ VmTP, const unsigned short* __restrict__ UTP,
    const float* __restrict__ drive_u, unsigned short* __restrict__ states_bf,
    unsigned short* __restrict__ slots_bf, unsigned short* __restrict__ heb,
    unsigned short* __restrict__ kc, unsigned short* __restrict__ vc) {
    extern __shared__ char smem_raw[];
    float* x = (float*)smem_raw;            // [16][512]
    float* drive = x + 8192;                // [16][512]
    float* buf1 = drive + 8192;             // wm_read / q / F
    float* buf2 = buf1 + 8192;              // hb_read
    float* sc = buf2 + 8192;                // [16][4][16]
    float* wt = sc + 1024;                  // [16][4][16]
    float* swm = wt + 1024;                 // [16][8]
    float* shb = swm + 128;                 // [16][32]
    float* acts = shb + 512;                // [16][8]
    int* idxm = (int*)(acts + 128);         // [16]
    unsigned short* xbf = (unsigned short*)(idxm + 16);  // [16][512] bf16
    unsigned short* abf = xbf + 8192;       // bf16 A staging [16][512]

    const int tid = threadIdx.x;
    const int wave = tid >> 5, lane = tid & 31;
    const float inv_sqd = 0.044194173824159216f;   // 1/sqrt(512)
    const float inv_sqdh = 0.08838834764831845f;   // 1/sqrt(128)

    // ---- zero-init all state (ws is poisoned by the harness) ----
    for (int i = tid; i < 8192; i += 1024) { x[i] = 0.f; xbf[i] = 0; }
    for (int i = tid; i < B_ * NSLOT_ * D_; i += 1024) slots_bf[i] = 0;
    for (int i = tid; i < HBROWS_ * B_ * D_; i += 1024) heb[i] = 0;
    for (int i = tid; i < CW_ * B_ * D_; i += 1024) { kc[i] = 0; vc[i] = 0; }
    if (tid < 128) acts[tid] = 0.f;
    __syncthreads();

    for (int t = 0; t < S_; ++t) {
        // ---- async-prefetch drive = u@Wu + bF for this step into LDS;
        //      overlaps with the score dots below, waited before first use ----
        for (int i = tid; i < 2048; i += 1024) {       // 2048 x 16B chunks
            int f = i << 2;                            // flat float index
            int b = f >> 9, d = f & 511;
            async_g2l_b128(drive + f, drive_u + (size_t)(b * S_ + t) * D_ + d);
        }
        // ---- working-memory scores: s_wm[b][n] = slots[b][n].x[b]/sqrt(d) ----
        {
            int p = tid >> 3, sub = tid & 7;   // 128 (b,n) pairs, 8 lanes each
            int b = p >> 3, n = p & 7;
            const unsigned short* sl = slots_bf + (size_t)(b * 8 + n) * D_ + sub * 64;
            const unsigned short* xr = xbf + b * D_ + sub * 64;
            float s = 0.f;
            for (int d = 0; d < 64; ++d) s += bf2f(sl[d]) * bf2f(xr[d]);
            s += __shfl_xor(s, 1, 32);
            s += __shfl_xor(s, 2, 32);
            s += __shfl_xor(s, 4, 32);
            if (sub == 0) swm[b * 8 + n] = s * inv_sqd;
        }
        // ---- Hebbian scores over the 32 live rows ----
        {
            int p = tid >> 1, sub = tid & 1;  // 512 (b,j) pairs
            int b = p >> 5, j = p & 31;
            const unsigned short* hr = heb + (size_t)(j * 16 + b) * D_ + sub * 256;
            const unsigned short* xr = xbf + b * D_ + sub * 256;
            float s = 0.f;
            for (int d = 0; d < 256; ++d) s += bf2f(hr[d]) * bf2f(xr[d]);
            s += __shfl_xor(s, 1, 32);
            if (sub == 0) shb[b * 32 + j] = s * inv_sqd;
        }
        __syncthreads();
        // ---- per-batch: top-3 softmax (wm), softmax w/ 96 implicit zero rows
        //      (hebbian), argmin of acts ----
        if (tid < 16) {
            int b = tid;
            float v[8];
            for (int n = 0; n < 8; ++n) v[n] = swm[b * 8 + n];
            int t0 = -1, t1 = -1; float bv; int bi; float kth;
            bi = -1; bv = -__builtin_inff();
            for (int n = 0; n < 8; ++n) if (v[n] > bv) { bv = v[n]; bi = n; }
            t0 = bi;
            bi = -1; bv = -__builtin_inff();
            for (int n = 0; n < 8; ++n) if (n != t0 && v[n] > bv) { bv = v[n]; bi = n; }
            t1 = bi;
            bi = -1; bv = -__builtin_inff();
            for (int n = 0; n < 8; ++n) if (n != t0 && n != t1 && v[n] > bv) { bv = v[n]; bi = n; }
            kth = bv;
            float mx = -__builtin_inff();
            for (int n = 0; n < 8; ++n) if (v[n] >= kth && v[n] > mx) mx = v[n];
            float den = 0.f, e[8];
            for (int n = 0; n < 8; ++n) {
                e[n] = (v[n] >= kth) ? __expf(v[n] - mx) : 0.f;
                den += e[n];
            }
            for (int n = 0; n < 8; ++n) swm[b * 8 + n] = e[n] / den;
        } else if (tid < 32) {
            int b = tid - 16;
            float mx = 0.f;   // zero rows have score 0
            for (int j = 0; j < 32; ++j) mx = fmaxf(mx, shb[b * 32 + j]);
            float den = 96.f * __expf(-mx);   // rows 32..127 of Hm are always 0
            for (int j = 0; j < 32; ++j) den += __expf(shb[b * 32 + j] - mx);
            for (int j = 0; j < 32; ++j)
                shb[b * 32 + j] = __expf(shb[b * 32 + j] - mx) / den;
        } else if (tid < 48) {
            int b = tid - 32;
            int bi = 0; float bv = acts[b * 8];
            for (int n = 1; n < 8; ++n)
                if (acts[b * 8 + n] < bv) { bv = acts[b * 8 + n]; bi = n; }
            idxm[b] = bi;
        }
        __syncthreads();
        // ---- reads + m = 0.5*(wm_read+hb_read) (bf16 A) ----
        for (int i = tid; i < 8192; i += 1024) {
            int b = i >> 9, d = i & 511;
            float wm = 0.f;
            for (int n = 0; n < 8; ++n)
                wm += swm[b * 8 + n] * bf2f(slots_bf[(size_t)(b * 8 + n) * D_ + d]);
            float hb = 0.f;
            for (int j = 0; j < 32; ++j)
                hb += shb[b * 32 + j] * bf2f(heb[(size_t)(j * 16 + b) * D_ + d]);
            abf[i] = f2bf(0.5f * (wm + hb));
        }
        __syncthreads();
        // ---- Hebbian store (every 4 steps) and slot write (both use pre-x) ----
        if ((t & 3) == 0) {
            int jr = t >> 2;
            for (int i = tid; i < 8192; i += 1024) {
                int b = i >> 9, d = i & 511;
                heb[(size_t)(jr * 16 + b) * D_ + d] = xbf[i];
            }
        }
        for (int i = tid; i < 8192; i += 1024) {
            int b = i >> 9, d = i & 511;
            slots_bf[(size_t)(b * 8 + idxm[b]) * D_ + d] = xbf[i];
        }
        if (tid < 16) {
            int b = tid;
            for (int n = 0; n < 8; ++n) acts[b * 8 + n] *= 0.95f;
            acts[b * 8 + idxm[b]] = 1.0f;
        }
        wait_async0();        // drive fully landed in LDS
        __syncthreads();
        // ---- q = x@Wq -> buf1 ; drive += m@Wmem  (wave = N-tile, 32 tiles) --
        {
            v8f a0 = {0.f, 0.f, 0.f, 0.f, 0.f, 0.f, 0.f, 0.f};
            v8f a1 = {0.f, 0.f, 0.f, 0.f, 0.f, 0.f, 0.f, 0.f};
            a0 = wmma_acc_tile(xbf, D_, WqP, D_, wave, a0);
            a1 = wmma_acc_tile(abf, D_, WmemP, D_, wave, a1);
            __syncthreads();   // all abf reads done before reuse
            int cn = lane & 15, hi = lane >> 4, col = wave * 16 + cn;
#pragma unroll
            for (int r = 0; r < 8; ++r) {
                int mr = r + hi * 8;
                buf1[mr * D_ + col] = a0[r];
                drive[mr * D_ + col] += a1[r];
            }
        }
        __syncthreads();
        // ---- multi-head attention over cached k/v ring ----
        {
            int b = tid >> 6, h = (tid >> 4) & 3, w = tid & 15;
            const float* qh = buf1 + b * D_ + h * DH_;
            const unsigned short* kr = kc + (size_t)(w * 16 + b) * D_ + h * DH_;
            float s = 0.f;
            for (int e = 0; e < DH_; ++e) s += qh[e] * bf2f(kr[e]);
            sc[tid] = s * inv_sqdh;   // sc[b][h][w]
        }
        __syncthreads();
        if (tid < 64) {
            int b = tid >> 2, h = tid & 3;
            float* p = sc + b * 64 + h * 16;
            float mx = p[0];
            for (int w = 1; w < 16; ++w) mx = fmaxf(mx, p[w]);
            float den = 0.f;
            for (int w = 0; w < 16; ++w) den += __expf(p[w] - mx);
            for (int w = 0; w < 16; ++w)
                wt[b * 64 + h * 16 + w] = __expf(p[w] - mx) / den;
        }
        __syncthreads();
        for (int i = tid; i < 8192; i += 1024) {   // attn_out -> abf (bf16)
            int b = i >> 9, d = i & 511, h = d >> 7;
            float s = 0.f;
            for (int w = 0; w < 16; ++w)
                s += wt[b * 64 + h * 16 + w] * bf2f(vc[(size_t)(w * 16 + b) * D_ + d]);
            abf[i] = f2bf(s);
        }
        __syncthreads();
        // ---- c = attn@Wo (bf16 back into abf), then drive += c@Wc ----
        {
            v8f a0 = {0.f, 0.f, 0.f, 0.f, 0.f, 0.f, 0.f, 0.f};
            a0 = wmma_acc_tile(abf, D_, WoP, D_, wave, a0);
            __syncthreads();
            int cn = lane & 15, hi = lane >> 4, col = wave * 16 + cn;
#pragma unroll
            for (int r = 0; r < 8; ++r) abf[(r + hi * 8) * D_ + col] = f2bf(a0[r]);
        }
        __syncthreads();
        {
            v8f a0 = {0.f, 0.f, 0.f, 0.f, 0.f, 0.f, 0.f, 0.f};
            a0 = wmma_acc_tile(abf, D_, WcP, D_, wave, a0);
            __syncthreads();
            int cn = lane & 15, hi = lane >> 4, col = wave * 16 + cn;
#pragma unroll
            for (int r = 0; r < 8; ++r) drive[(r + hi * 8) * D_ + col] += a0[r];
        }
        __syncthreads();
        // ---- 4 Euler substeps: F = tanh(x@Vm^T)@U^T + drive ----
        for (int ss = 0; ss < 4; ++ss) {
            if (wave < 8) {   // h = tanh(x@Vm^T): N=128 -> 8 tiles
                v8f a0 = {0.f, 0.f, 0.f, 0.f, 0.f, 0.f, 0.f, 0.f};
                a0 = wmma_acc_tile(xbf, D_, VmTP, D_, wave, a0);
                int cn = lane & 15, hi = lane >> 4, col = wave * 16 + cn;
#pragma unroll
                for (int r = 0; r < 8; ++r)
                    abf[(r + hi * 8) * R_ + col] = f2bf(tanhf(a0[r]));
            }
            __syncthreads();
            {   // F = h@U^T + drive -> buf1 (K=128)
                v8f a0 = {0.f, 0.f, 0.f, 0.f, 0.f, 0.f, 0.f, 0.f};
                a0 = wmma_acc_tile(abf, R_, UTP, R_, wave, a0);
                int cn = lane & 15, hi = lane >> 4, col = wave * 16 + cn;
#pragma unroll
                for (int r = 0; r < 8; ++r) {
                    int mr = r + hi * 8;
                    buf1[mr * D_ + col] = a0[r] + drive[mr * D_ + col];
                }
            }
            __syncthreads();
            for (int i = tid; i < 8192; i += 1024) {
                int d = i & 511;
                float tau = (d < 256) ? 0.1f : 1.0f;   // FAST*d = 256
                float xn = x[i] + 0.25f * ((-0.1f * x[i] + buf1[i]) / tau);
                x[i] = xn;
                xbf[i] = f2bf(xn);
            }
            __syncthreads();
        }
        // ---- epilogue: record state; refresh k/v cache row t%16 ----
        for (int i = tid; i < 8192; i += 1024) {
            int b = i >> 9, d = i & 511;
            states_bf[(size_t)(b * S_ + t) * D_ + d] = xbf[i];
        }
        {
            v8f a0 = {0.f, 0.f, 0.f, 0.f, 0.f, 0.f, 0.f, 0.f};
            v8f a1 = {0.f, 0.f, 0.f, 0.f, 0.f, 0.f, 0.f, 0.f};
            a0 = wmma_acc_tile(xbf, D_, WkP, D_, wave, a0);
            a1 = wmma_acc_tile(xbf, D_, WvP, D_, wave, a1);
            int cn = lane & 15, hi = lane >> 4, col = wave * 16 + cn;
            int ws = t & 15;
#pragma unroll
            for (int r = 0; r < 8; ++r) {
                int b = r + hi * 8;
                kc[(size_t)(ws * 16 + b) * D_ + col] = f2bf(a0[r]);
                vc[(size_t)(ws * 16 + b) * D_ + col] = f2bf(a1[r]);
            }
        }
        __syncthreads();
    }
}

// ---------------------------------------------------------------------------
extern "C" void kernel_launch(void* const* d_in, const int* in_sizes, int n_in,
                              void* d_out, int out_size, void* d_ws, size_t ws_size,
                              hipStream_t stream) {
    const int* ids = (const int*)d_in[0];
    const float* embed = (const float*)d_in[1];
    const float* Wq = (const float*)d_in[2];
    const float* Wk = (const float*)d_in[3];
    const float* Wv = (const float*)d_in[4];
    const float* Wo = (const float*)d_in[5];
    const float* U = (const float*)d_in[6];
    const float* Vm = (const float*)d_in[7];
    const float* Wu = (const float*)d_in[8];
    const float* Wmem = (const float*)d_in[9];
    const float* Wc = (const float*)d_in[10];
    const float* bF = (const float*)d_in[11];
    const float* Wout = (const float*)d_in[12];
    const float* bout = (const float*)d_in[13];

    char* ws = (char*)d_ws;
    size_t off = 0;
    auto carve = [&](size_t bytes) {
        void* p = ws + off;
        off += (bytes + 255) & ~(size_t)255;
        return p;
    };
    const size_t W512 = (size_t)512 * 512 * 2;
    unsigned short* WqP = (unsigned short*)carve(W512);
    unsigned short* WkP = (unsigned short*)carve(W512);
    unsigned short* WvP = (unsigned short*)carve(W512);
    unsigned short* WoP = (unsigned short*)carve(W512);
    unsigned short* WmemP = (unsigned short*)carve(W512);
    unsigned short* WcP = (unsigned short*)carve(W512);
    unsigned short* WuP = (unsigned short*)carve(W512);
    unsigned short* VmTP = (unsigned short*)carve((size_t)512 * 128 * 2);
    unsigned short* UTP = (unsigned short*)carve((size_t)128 * 512 * 2);
    unsigned short* WoutP = (unsigned short*)carve((size_t)512 * 32000 * 2);
    float* drive_u = (float*)carve((size_t)2048 * 512 * 4);
    unsigned short* states_bf = (unsigned short*)carve((size_t)2048 * 512 * 2);
    unsigned short* slots_bf = (unsigned short*)carve((size_t)16 * 8 * 512 * 2);
    unsigned short* heb = (unsigned short*)carve((size_t)32 * 16 * 512 * 2);
    unsigned short* kc = (unsigned short*)carve((size_t)16 * 16 * 512 * 2);
    unsigned short* vc = (unsigned short*)carve((size_t)16 * 16 * 512 * 2);

    // ---- pack all matmul operands to bf16 WMMA-B layout ----
    pack_b_kernel<<<256, 256, 0, stream>>>(Wq, WqP, 512, 512, 0);
    pack_b_kernel<<<256, 256, 0, stream>>>(Wk, WkP, 512, 512, 0);
    pack_b_kernel<<<256, 256, 0, stream>>>(Wv, WvP, 512, 512, 0);
    pack_b_kernel<<<256, 256, 0, stream>>>(Wo, WoP, 512, 512, 0);
    pack_b_kernel<<<256, 256, 0, stream>>>(Wmem, WmemP, 512, 512, 0);
    pack_b_kernel<<<256, 256, 0, stream>>>(Wc, WcP, 512, 512, 0);
    pack_b_kernel<<<256, 256, 0, stream>>>(Wu, WuP, 512, 512, 0);
    pack_b_kernel<<<128, 256, 0, stream>>>(Vm, VmTP, 512, 128, 1);  // Vm^T
    pack_b_kernel<<<128, 256, 0, stream>>>(U, UTP, 128, 512, 1);    // U^T
    pack_b_kernel<<<4096, 256, 0, stream>>>(Wout, WoutP, 512, 32000, 0);

    // ---- phase 1: drive_u = embed[ids] @ Wu + bF  (gathered WMMA GEMM) ----
    {
        dim3 grid(2048 / 64, 512 / 32);
        gemm_bf16_kernel<<<grid, 256, 64 * 512 * 2, stream>>>(
            nullptr, ids, embed, WuP, bF, drive_u, 512, 512, 1);
    }
    // ---- phase 2: serial recurrence, one 32-wave workgroup ----
    {
        size_t smem = (size_t)(8192 * 4) * 4      // x, drive, buf1, buf2
                    + (1024 + 1024 + 128 + 512 + 128 + 16) * 4  // sc,wt,swm,shb,acts,idxm
                    + (size_t)8192 * 2 * 2;       // xbf, abf
        recurrence_kernel<<<1, 1024, smem, stream>>>(
            WqP, WkP, WvP, WoP, WmemP, WcP, VmTP, UTP,
            drive_u, states_bf, slots_bf, heb, kc, vc);
    }
    // ---- phase 3: out = states @ Wout + bout  (dominant WMMA GEMM) ----
    {
        dim3 grid(2048 / 64, 32000 / 32);
        gemm_bf16_kernel<<<grid, 256, 64 * 512 * 2, stream>>>(
            states_bf, nullptr, nullptr, WoutP, bout, (float*)d_out, 32000, 512, 0);
    }
}
